// SparseTemporalFusion_55465207660767
// MI455X (gfx1250) — compile-verified, matrix operand order checked by reading, not run on hardware
//
#include <hip/hip_runtime.h>

// ---------------------------------------------------------------------------
// Problem constants (from the reference)
// ---------------------------------------------------------------------------
constexpr int C    = 384;
constexpr int NH   = 8;
constexpr int DH   = 48;          // head dim
constexpr int T    = 3;
constexpr int G    = 160;         // grid side (320/2)
constexpr int NT   = G * G;       // 25600 tokens per frame
constexpr int N    = T * NT;      // 76800 tokens
constexpr int NW   = 33;
constexpr int NWIN = NW * NW;     // 1089 windows
constexpr int CAP  = 80;
constexpr int ROW0 = N - NT;      // 51200: first output row

// CDNA5 WMMA vector types (wave32)
typedef __bf16 v16bf  __attribute__((ext_vector_type(16)));
typedef float  v8f    __attribute__((ext_vector_type(8)));
typedef __bf16 bf16x8 __attribute__((ext_vector_type(8)));

// ---------------------------------------------------------------------------
// Fragment loader for V_WMMA_F32_16X16X32_BF16.
// Per CDNA5 ISA 16-bit A-matrix 16x32 layout: lane L holds row (L&15);
// lanes 0-15 carry K = {0..7, 16..23}, lanes 16-31 carry K = {8..15, 24..31}.
// B operands are stored N-major ("transposed") so the same loader serves both.
// ---------------------------------------------------------------------------
__device__ inline v16bf load_frag(const __bf16* base, int stride) {
  const int lane = threadIdx.x & 31;
  const __bf16* p = base + (lane & 15) * stride + ((lane >> 4) << 3);
  v16bf f;
#pragma unroll
  for (int i = 0; i < 8; ++i) {
    f[i]     = p[i];
    f[i + 8] = p[16 + i];
  }
  return f;
}

__device__ inline v8f wmma_bf16(v16bf a, v16bf b, v8f c) {
  return __builtin_amdgcn_wmma_f32_16x16x32_bf16(
      /*neg_a=*/false, a, /*neg_b=*/false, b,
      /*c_mod=*/(short)0, c, /*reuse_a=*/false, /*reuse_b=*/false);
}

// CDNA5 async global->LDS copy (16 bytes per lane), tracked by ASYNCcnt.
__device__ inline void async_b128(unsigned lds_off, const __bf16* gaddr) {
  asm volatile("global_load_async_to_lds_b128 %0, %1, off"
               :: "v"(lds_off), "v"(gaddr) : "memory");
}

// ---------------------------------------------------------------------------
// prep: x = feats + time_emb[frame_ids]; emit f32 + bf16 copies
// ---------------------------------------------------------------------------
__global__ __launch_bounds__(256) void k_prep(
    const float* __restrict__ feats, const int* __restrict__ fids,
    const float* __restrict__ temb, float* __restrict__ x,
    __bf16* __restrict__ xb) {
  int i = blockIdx.x * 256 + threadIdx.x;
  if (i >= N * C) return;
  int row = i / C, col = i - row * C;
  float v = feats[i] + temb[fids[row] * C + col];
  x[i]  = v;
  xb[i] = (__bf16)v;
}

// Weight prep: f32 KxNn -> bf16 transposed NnxK (dst[n*K+k] = src[k*Nn+n]).
// Lets GEMM stage B tiles with contiguous async b128 copies (no transpose).
__global__ __launch_bounds__(256) void k_cvt_t(
    const float* __restrict__ src, __bf16* __restrict__ dst, int K, int Nn) {
  int i = blockIdx.x * 256 + threadIdx.x;
  if (i >= K * Nn) return;
  int n = i / K, k = i - n * K;
  dst[i] = (__bf16)src[(size_t)k * Nn + n];
}

// ---------------------------------------------------------------------------
// Analytic window membership (deterministic, matches stable argsort order)
// ---------------------------------------------------------------------------
__global__ __launch_bounds__(64) void k_windows(
    int* __restrict__ wcnt, int* __restrict__ wmem) {
  int w = blockIdx.x * 64 + threadIdx.x;
  if (w >= NWIN) return;
  int wy = w / NW, wx = w - wy * NW;
  int gy0 = max(5 * wy - 2, 0), gy1 = min(5 * wy + 2, G - 1);
  int gx0 = max(5 * wx - 2, 0), gx1 = min(5 * wx + 2, G - 1);
  int c = 0;
  for (int t = 0; t < T; ++t)
    for (int gy = gy0; gy <= gy1; ++gy)
      for (int gx = gx0; gx <= gx1; ++gx)
        wmem[w * CAP + c++] = t * NT + gy * G + gx;
  wcnt[w] = c;
}

// ---------------------------------------------------------------------------
// Tiled WMMA GEMM with double-buffered async-LDS staging.
// out[M,Nout] = A[M,K] * Bt[Nout,K]^T  (+bias, epilogue EPI)
//   EPI 0: +bias            -> bf16
//   EPI 1: +bias            -> f32
//   EPI 2: +bias, relu      -> bf16
//   EPI 3: +bias, +residual -> f32
// Block tile 128x64, BK=64, 8 waves; wave w owns a 16x64 strip (4 accums).
// Per stage each thread issues 6 global_load_async_to_lds_b128 (4 A + 2 B);
// s_wait_asynccnt pipelines stage k+1 staging under stage k compute.
// M % 128 == 0, Nout % 64 == 0, K % 64 == 0 (all true here).
// ---------------------------------------------------------------------------
template <int EPI>
__global__ __launch_bounds__(256) void k_gemm(
    const __bf16* __restrict__ A, const __bf16* __restrict__ Bt,
    const float* __restrict__ bias, const float* __restrict__ residual,
    void* __restrict__ out, int M, int K, int Nout) {
  constexpr int BM = 128, BN = 64, BK = 64;
  constexpr int STR = 72;                    // padded LDS row stride (bf16)
  __shared__ __bf16 As[2][BM * STR];
  __shared__ __bf16 Bs[2][BN * STR];

  const int bm = blockIdx.x * BM;
  const int bn = blockIdx.y * BN;
  const int tid  = threadIdx.x;
  const int wave = tid >> 5;
  const int lane = tid & 31;
  const int wm = wave * 16;

  v8f acc[4] = {{}, {}, {}, {}};

  auto issue_stage = [&](int stage, int k0) {
    unsigned abase = (unsigned)(size_t)(const void*)&As[stage][0];
    unsigned bbase = (unsigned)(size_t)(const void*)&Bs[stage][0];
#pragma unroll
    for (int i = 0; i < 4; ++i) {            // A: 128x64 = 1024 b128 chunks
      int q = tid + i * 256, r = q >> 3, c8 = (q & 7) * 8;
      async_b128(abase + (unsigned)(r * STR + c8) * 2,
                 A + (size_t)(bm + r) * K + k0 + c8);
    }
#pragma unroll
    for (int i = 0; i < 2; ++i) {            // B: 64x64 = 512 b128 chunks
      int q = tid + i * 256, r = q >> 3, c8 = (q & 7) * 8;
      async_b128(bbase + (unsigned)(r * STR + c8) * 2,
                 Bt + (size_t)(bn + r) * K + k0 + c8);
    }
  };

  issue_stage(0, 0);
  const int nk = K / BK;
  for (int ks = 0; ks < nk; ++ks) {
    if (ks + 1 < nk) {
      issue_stage((ks + 1) & 1, (ks + 1) * BK);
      asm volatile("s_wait_asynccnt 6" ::: "memory");  // stage ks landed
    } else {
      asm volatile("s_wait_asynccnt 0" ::: "memory");
    }
    __syncthreads();

    const __bf16* Ab = &As[ks & 1][0];
    const __bf16* Bb = &Bs[ks & 1][0];
#pragma unroll
    for (int kc = 0; kc < BK; kc += 32) {
      v16bf a = load_frag(Ab + wm * STR + kc, STR);
#pragma unroll
      for (int j = 0; j < 4; ++j) {
        v16bf b = load_frag(Bb + j * 16 * STR + kc, STR);
        acc[j] = wmma_bf16(a, b, acc[j]);
      }
    }
    __syncthreads();   // reads done before this buffer is re-staged
  }

  // epilogue: C-layout -> lanes 0-15: M=r, lanes 16-31: M=r+8; N = lane&15
  __bf16* outb = (__bf16*)out;
  float*  outf = (float*)out;
#pragma unroll
  for (int r = 0; r < 8; ++r) {
    const int m  = bm + wm + r + ((lane >> 4) << 3);
    const int nl = lane & 15;
#pragma unroll
    for (int j = 0; j < 4; ++j) {
      const int n = bn + j * 16 + nl;
      float val = acc[j][r] + bias[n];
      if constexpr (EPI == 2) val = fmaxf(val, 0.0f);
      if constexpr (EPI == 3) val += residual[(size_t)m * Nout + n];
      const size_t off = (size_t)m * Nout + n;
      if constexpr (EPI == 0 || EPI == 2) outb[off] = (__bf16)val;
      else                                outf[off] = val;
    }
  }
}

// ---------------------------------------------------------------------------
// Windowed attention: one block (256 thr / 8 waves) per window.
// ---------------------------------------------------------------------------
constexpr int KD = 64;   // padded head dim
constexpr int KT = 96;   // padded token dim

__global__ __launch_bounds__(256) void k_attn(
    const __bf16* __restrict__ qkv, const int* __restrict__ wcnt,
    const int* __restrict__ wmem, __bf16* __restrict__ attn_out /* NT x C */) {
  __shared__ __bf16 Qs[CAP * KD];
  __shared__ __bf16 Ks[CAP * KD];
  __shared__ __bf16 Vt[DH * KT];     // V transposed: [d][token]
  __shared__ float  Sf[CAP * CAP];
  __shared__ __bf16 Pm[CAP * KT];    // softmax probs, token dim padded
  __shared__ int    mem[CAP];

  const int w    = blockIdx.x;
  const int tid  = threadIdx.x;
  const int wave = tid >> 5;
  const int lane = tid & 31;
  const int cnt  = wcnt[w];
  const float scale = rsqrtf((float)DH);

  // zero padded operand buffers once; cnt is constant per window, so the
  // pad regions (rows/cols >= cnt, head-dim cols >= 48, Pm cols >= 80)
  // stay exactly zero across all heads.
  for (int i = tid; i < CAP * KD / 2; i += 256) ((int*)Qs)[i] = 0;
  for (int i = tid; i < CAP * KD / 2; i += 256) ((int*)Ks)[i] = 0;
  for (int i = tid; i < DH * KT / 2;  i += 256) ((int*)Vt)[i] = 0;
  for (int i = tid; i < CAP * KT / 2; i += 256) ((int*)Pm)[i] = 0;
  if (tid < CAP) mem[tid] = (tid < cnt) ? wmem[w * CAP + tid] : 0;
  __syncthreads();

  for (int h = 0; h < NH; ++h) {
    // vectorized gather: 6 x b128 per member row per operand (48 bf16)
    for (int i = tid; i < cnt * 6; i += 256) {
      int row = i / 6, c8 = (i - row * 6) * 8;
      const __bf16* src = qkv + (size_t)mem[row] * (3 * C) + h * DH + c8;
      bf16x8 qv = *(const bf16x8*)(src);
      bf16x8 kv = *(const bf16x8*)(src + C);
      bf16x8 vv = *(const bf16x8*)(src + 2 * C);
      *(bf16x8*)(Qs + row * KD + c8) = qv;
      *(bf16x8*)(Ks + row * KD + c8) = kv;
#pragma unroll
      for (int e = 0; e < 8; ++e) Vt[(c8 + e) * KT + row] = vv[e];
    }
    __syncthreads();

    // S = Q K^T  (both operands row-major per-lane-row -> same frag loader)
    for (int tile = wave; tile < 25; tile += 8) {
      int tm = (tile / 5) * 16, tn = (tile % 5) * 16;
      v8f acc = {};
#pragma unroll
      for (int k0 = 0; k0 < KD; k0 += 32) {
        v16bf a = load_frag(Qs + tm * KD + k0, KD);
        v16bf b = load_frag(Ks + tn * KD + k0, KD);
        acc = wmma_bf16(a, b, acc);
      }
#pragma unroll
      for (int r = 0; r < 8; ++r) {
        int m = tm + r + ((lane >> 4) << 3);
        int n = tn + (lane & 15);
        Sf[m * CAP + n] = acc[r];
      }
    }
    __syncthreads();

    // softmax over the cnt valid keys (invalid keys contribute exactly 0 in
    // the reference: exp(-1e9 - mx) underflows; Pm cols >= cnt stay zero).
    if (tid < CAP) {
      float* srow = Sf + tid * CAP;
      float mx = -1e30f;
      for (int j = 0; j < cnt; ++j) {
        float s = srow[j] * scale;
        srow[j] = s;
        mx = fmaxf(mx, s);
      }
      float sum = 0.0f;
      for (int j = 0; j < cnt; ++j) {
        float e = __expf(srow[j] - mx);
        srow[j] = e;
        sum += e;
      }
      float inv = 1.0f / sum;
      for (int j = 0; j < cnt; ++j)
        Pm[tid * KT + j] = (__bf16)(srow[j] * inv);
    }
    __syncthreads();

    // O = P V  (P: [query][token], Vt: [d][token] -> B via same loader)
    for (int tile = wave; tile < 15; tile += 8) {
      int tm = (tile / 3) * 16, tn = (tile % 3) * 16;
      v8f acc = {};
#pragma unroll
      for (int k0 = 0; k0 < KT; k0 += 32) {
        v16bf a = load_frag(Pm + tm * KT + k0, KT);
        v16bf b = load_frag(Vt + tn * KT + k0, KT);
        acc = wmma_bf16(a, b, acc);
      }
#pragma unroll
      for (int r = 0; r < 8; ++r) {
        int m = tm + r + ((lane >> 4) << 3);
        int n = tn + (lane & 15);
        if (m < cnt) {
          int tok = mem[m];
          if (tok >= ROW0)  // only last-frame rows feed the output path
            attn_out[(size_t)(tok - ROW0) * C + h * DH + n] = (__bf16)acc[r];
        }
      }
    }
    __syncthreads();
  }
}

// ---------------------------------------------------------------------------
// Fused residual + LayerNorm; one wave per row (wave32 shuffles)
// ---------------------------------------------------------------------------
__global__ __launch_bounds__(256) void k_ln(
    const float* __restrict__ proj, const float* __restrict__ x,
    const float* __restrict__ g, const float* __restrict__ b,
    float* __restrict__ h, __bf16* __restrict__ hb) {
  const int row  = blockIdx.x * 8 + (threadIdx.x >> 5);
  const int lane = threadIdx.x & 31;
  const float* p  = proj + (size_t)row * C;
  const float* xr = x + (size_t)(row + ROW0) * C;

  float v[12];
  float s = 0.0f;
#pragma unroll
  for (int i = 0; i < 12; ++i) {
    int c = lane + i * 32;
    v[i] = p[c] + xr[c];
    s += v[i];
  }
#pragma unroll
  for (int o = 16; o > 0; o >>= 1) s += __shfl_xor(s, o, 32);
  const float mean = s / (float)C;

  float vs = 0.0f;
#pragma unroll
  for (int i = 0; i < 12; ++i) {
    float d = v[i] - mean;
    vs += d * d;
  }
#pragma unroll
  for (int o = 16; o > 0; o >>= 1) vs += __shfl_xor(vs, o, 32);
  const float rstd = rsqrtf(vs / (float)C + 1e-5f);

#pragma unroll
  for (int i = 0; i < 12; ++i) {
    int c = lane + i * 32;
    float o = (v[i] - mean) * rstd * g[c] + b[c];
    h[(size_t)row * C + c]  = o;
    hb[(size_t)row * C + c] = (__bf16)o;
  }
}

// ---------------------------------------------------------------------------
// Host side
// ---------------------------------------------------------------------------
static inline size_t align_up(size_t v, size_t a) { return (v + a - 1) / a * a; }

extern "C" void kernel_launch(void* const* d_in, const int* in_sizes, int n_in,
                              void* d_out, int out_size, void* d_ws, size_t ws_size,
                              hipStream_t stream) {
  (void)in_sizes; (void)n_in; (void)out_size; (void)ws_size;
  const float* feats  = (const float*)d_in[0];
  const int*   fids   = (const int*)d_in[2];
  const float* temb   = (const float*)d_in[3];
  const float* w_qkv  = (const float*)d_in[4];
  const float* b_qkv  = (const float*)d_in[5];
  const float* w_proj = (const float*)d_in[6];
  const float* b_proj = (const float*)d_in[7];
  const float* ln_g   = (const float*)d_in[8];
  const float* ln_b   = (const float*)d_in[9];
  const float* w1     = (const float*)d_in[10];
  const float* b1     = (const float*)d_in[11];
  const float* w2     = (const float*)d_in[12];
  const float* b2     = (const float*)d_in[13];

  // workspace layout (~515 MB)
  char* base = (char*)d_ws;
  size_t off = 0;
  auto alloc = [&](size_t bytes) {
    void* p = base + off;
    off = align_up(off + bytes, 256);
    return p;
  };
  float*  x_f32   = (float*) alloc((size_t)N * C * 4);
  __bf16* x_bf    = (__bf16*)alloc((size_t)N * C * 2);
  __bf16* qkv_bf  = (__bf16*)alloc((size_t)N * 3 * C * 2);
  __bf16* wqkv_t  = (__bf16*)alloc((size_t)C * 3 * C * 2);   // [3C][C]
  __bf16* wproj_t = (__bf16*)alloc((size_t)C * C * 2);       // [C][C]
  __bf16* w1_t    = (__bf16*)alloc((size_t)C * 2 * C * 2);   // [2C][C]
  __bf16* w2_t    = (__bf16*)alloc((size_t)2 * C * C * 2);   // [C][2C]
  int*    wcnt    = (int*)   alloc((size_t)NWIN * 4);
  int*    wmem    = (int*)   alloc((size_t)NWIN * CAP * 4);
  __bf16* attn_bf = (__bf16*)alloc((size_t)NT * C * 2);
  float*  proj_f  = (float*) alloc((size_t)NT * C * 4);
  float*  h_f32   = (float*) alloc((size_t)NT * C * 4);
  __bf16* h_bf    = (__bf16*)alloc((size_t)NT * C * 2);
  __bf16* hid_bf  = (__bf16*)alloc((size_t)NT * 2 * C * 2);

  // 1. x = feats + time_emb[fid]
  k_prep<<<(N * C + 255) / 256, 256, 0, stream>>>(feats, fids, temb, x_f32, x_bf);

  // 2. weight conversions f32 -> bf16, transposed to N-major
  k_cvt_t<<<(C * 3 * C + 255) / 256, 256, 0, stream>>>(w_qkv, wqkv_t, C, 3 * C);
  k_cvt_t<<<(C * C + 255) / 256, 256, 0, stream>>>(w_proj, wproj_t, C, C);
  k_cvt_t<<<(C * 2 * C + 255) / 256, 256, 0, stream>>>(w1, w1_t, C, 2 * C);
  k_cvt_t<<<(2 * C * C + 255) / 256, 256, 0, stream>>>(w2, w2_t, 2 * C, C);

  // 3. window membership (analytic, deterministic)
  k_windows<<<(NWIN + 63) / 64, 64, 0, stream>>>(wcnt, wmem);

  // 4. qkv = x @ w_qkv + b_qkv  (76800 x 384 x 1152) -> bf16
  k_gemm<0><<<dim3(N / 128, 3 * C / 64), 256, 0, stream>>>(
      x_bf, wqkv_t, b_qkv, nullptr, qkv_bf, N, C, 3 * C);

  // 5. windowed attention -> attn_bf (last-frame rows only)
  k_attn<<<NWIN, 256, 0, stream>>>(qkv_bf, wcnt, wmem, attn_bf);

  // 6. proj (25600 x 384 x 384) -> f32
  k_gemm<1><<<dim3(NT / 128, C / 64), 256, 0, stream>>>(
      attn_bf, wproj_t, b_proj, nullptr, proj_f, NT, C, C);

  // 7. residual + LayerNorm
  k_ln<<<NT / 8, 256, 0, stream>>>(proj_f, x_f32, ln_g, ln_b, h_f32, h_bf);

  // 8. MLP fc1 + relu (25600 x 384 x 768) -> bf16
  k_gemm<2><<<dim3(NT / 128, 2 * C / 64), 256, 0, stream>>>(
      h_bf, w1_t, b1, nullptr, hid_bf, NT, C, 2 * C);

  // 9. MLP fc2 + residual (25600 x 768 x 384) -> d_out (f32)
  k_gemm<3><<<dim3(NT / 128, C / 64), 256, 0, stream>>>(
      hid_bf, w2_t, b2, h_f32, d_out, NT, 2 * C, C);
}
